// SelfSelfAttention_38208029065586
// MI455X (gfx1250) — matrix operand order repaired and checked by reading, not compiled, not run on hardware
//
#include <hip/hip_runtime.h>
#include <math.h>
#include <stdint.h>

// ---------------- constants ----------------
constexpr int BQ   = 8;     // batch
constexpr int NN   = 577;   // valid sequence length
constexpr int NPAD = 640;   // padded N (multiple of 64)
constexpr int CC   = 1024;  // channels
constexpr int HH   = 16;    // heads
constexpr int HD   = 64;    // head dim
constexpr float SCALE = 0.125f; // hd^-0.5

constexpr int LDG = 40;  // LDS row stride (elems) for 32-wide GEMM tiles (bank-safe)
constexpr int LDK = 72;  // LDS row stride (elems) for 64-wide K/V tiles (bank-safe)

#define DEVI __device__ __forceinline__

typedef __attribute__((ext_vector_type(16))) __bf16 v16bf;
typedef __attribute__((ext_vector_type(8)))  __bf16 v8bf;
typedef __attribute__((ext_vector_type(8)))  float  v8f;

DEVI v8f v8zero() {
  v8f z;
#pragma unroll
  for (int i = 0; i < 8; ++i) z[i] = 0.0f;
  return z;
}

// A fragment (16x32 bf16), row-major source with leading dim ld (elements).
DEVI v16bf ldA(const __bf16* base, int ld, int lane) {
  int m = lane & 15, hi = lane >> 4;
  const __bf16* p = base + (size_t)m * ld + hi * 16;
  v8bf lo = *(const v8bf*)(p);
  v8bf hh = *(const v8bf*)(p + 8);
  v16bf r;
#pragma unroll
  for (int e = 0; e < 8; ++e) { r[e] = lo[e]; r[e + 8] = hh[e]; }
  return r;
}

// B fragment (32x16 bf16) where B[k][n] = src[n][k] (src row-major, ld elements).
DEVI v16bf ldB(const __bf16* base, int ld, int lane) {
  int n = lane & 15, hi = lane >> 4;
  const __bf16* p = base + (size_t)n * ld + hi * 16;
  v8bf lo = *(const v8bf*)(p);
  v8bf hh = *(const v8bf*)(p + 8);
  v16bf r;
#pragma unroll
  for (int e = 0; e < 8; ++e) { r[e] = lo[e]; r[e + 8] = hh[e]; }
  return r;
}

// B fragment for P@V from a linear [32 keys][LDK] LDS tile: B[k][n] = V[k][n]
DEVI v16bf ldBVlin(const __bf16* vb, int lane, int sub) {
  int n = sub * 16 + (lane & 15);
  int kb = (lane >> 4) * 16;
  v16bf r;
#pragma unroll
  for (int e = 0; e < 16; ++e) r[e] = vb[(kb + e) * LDK + n];
  return r;
}

DEVI v8f wmma_bf(v16bf a, v16bf b, v8f c) {
  return __builtin_amdgcn_wmma_f32_16x16x32_bf16(false, a, false, b, (short)0, c,
                                                 false, false);
}

// reductions across the 16 lanes of a row group (do NOT cross lane-16 boundary)
DEVI float rmax16(float v) {
#pragma unroll
  for (int m = 1; m < 16; m <<= 1) v = fmaxf(v, __shfl_xor(v, m, 32));
  return v;
}
DEVI float rsum16(float v) {
#pragma unroll
  for (int m = 1; m < 16; m <<= 1) v += __shfl_xor(v, m, 32);
  return v;
}

// async 16B copy global -> LDS (ASYNCcnt-tracked, CDNA5 path)
DEVI void async_cp16(unsigned lds_off, const char* gsrc) {
  asm volatile("global_load_async_to_lds_b128 %0, %1, off"
               :: "v"(lds_off), "v"(gsrc)
               : "memory");
}
DEVI void wait_async0() { asm volatile("s_wait_asynccnt 0x0" ::: "memory"); }

// ---------------- prep: x[N,B,C] fp32 -> xb[B,NPAD,C] bf16 + row norms ----------------
__global__ void prep_kernel(const float* __restrict__ x, __bf16* __restrict__ xb,
                            float* __restrict__ rnorm) {
  int n = blockIdx.x, b = blockIdx.y, t = threadIdx.x;
  __bf16* dst = xb + ((size_t)b * NPAD + n) * CC;
  if (n >= NN) {  // zero pad rows (whole block takes this branch uniformly)
    for (int c = t; c < CC; c += 256) dst[c] = (__bf16)0.0f;
    return;
  }
  const float* src = x + ((size_t)n * BQ + b) * CC;
  float ss = 0.0f;
  for (int c = t; c < CC; c += 256) {
    float v = src[c];
    dst[c] = (__bf16)v;
    ss += v * v;
  }
  __shared__ float red[8];
#pragma unroll
  for (int m = 1; m < 32; m <<= 1) ss += __shfl_xor(ss, m, 32);
  if ((t & 31) == 0) red[t >> 5] = ss;
  __syncthreads();
  if (t == 0) {
    float s = 0.0f;
#pragma unroll
    for (int i = 0; i < 8; ++i) s += red[i];
    rnorm[(size_t)b * NN + n] = sqrtf(s);
  }
}

// ---------------- inv_temp[b] = mean(||row||) * scale (deterministic) ----------------
__global__ void fin_kernel(const float* __restrict__ rnorm, float* __restrict__ itemp) {
  __shared__ float sh[256];
  int b = blockIdx.x, t = threadIdx.x;
  float s = 0.0f;
  for (int n = t; n < NN; n += 256) s += rnorm[(size_t)b * NN + n];
  sh[t] = s;
  __syncthreads();
  for (int off = 128; off > 0; off >>= 1) {
    if (t < off) sh[t] += sh[t + off];
    __syncthreads();
  }
  if (t == 0) itemp[b] = sh[0] * (SCALE / (float)NN);
}

// ---------------- weight fp32 -> bf16 ----------------
__global__ void convw_kernel(const float* __restrict__ qkvw, const float* __restrict__ projw,
                             __bf16* __restrict__ Wq, __bf16* __restrict__ Wp) {
  const size_t n1 = (size_t)3 * CC * CC, n2 = (size_t)CC * CC;
  for (size_t i = (size_t)blockIdx.x * blockDim.x + threadIdx.x; i < n1 + n2;
       i += (size_t)gridDim.x * blockDim.x) {
    if (i < n1) Wq[i] = (__bf16)qkvw[i];
    else        Wp[i - n1] = (__bf16)projw[i - n1];
  }
}

// ======== shared GEMM core: 64x64 block tile, async-LDS double buffered k-loop ========
// As/Bs: [2][64*LDG] LDS tiles. Ag: A rows (block M base), Wg: W rows (block N base).
DEVI void gemm_core(const __bf16* Ag, const __bf16* Wg, int lane, int wv, int tid,
                    __bf16 (*As)[64 * LDG], __bf16 (*Bs)[64 * LDG], v8f acc[4]) {
  unsigned aOff[2], bOff[2];
#pragma unroll
  for (int i = 0; i < 2; ++i) {
    aOff[i] = (unsigned)(size_t)(void*)&As[i][0];
    bOff[i] = (unsigned)(size_t)(void*)&Bs[i][0];
  }
  // stage one 64x32 A tile + 64x32 B tile: 256+256 16B chunks, 4 per thread
  auto stage = [&](int p, int k0) {
    const char* ag = (const char*)(Ag + k0);
    const char* wg = (const char*)(Wg + k0);
#pragma unroll
    for (int i = 0; i < 2; ++i) {
      int c = tid * 2 + i;  // 0..255
      unsigned row = (unsigned)(c >> 2), seg = (unsigned)(c & 3);
      unsigned loff = row * (LDG * 2) + seg * 16;
      size_t goff = (size_t)row * (CC * 2) + seg * 16;
      async_cp16(aOff[p] + loff, ag + goff);
      async_cp16(bOff[p] + loff, wg + goff);
    }
  };

  stage(0, 0);
  wait_async0();
  __syncthreads();
  int p = 0;
  for (int k0 = 0; k0 < CC; k0 += 32) {
    if (k0 + 32 < CC) stage(1 - p, k0 + 32);  // prefetch next k-slab while computing
    const __bf16* Al = As[p];
    const __bf16* Bl = Bs[p];
    v16bf af = ldA(Al + wv * 16 * LDG, LDG, lane);
#pragma unroll
    for (int s = 0; s < 4; ++s)
      acc[s] = wmma_bf(af, ldB(Bl + s * 16 * LDG, LDG, lane), acc[s]);
    wait_async0();
    __syncthreads();
    p ^= 1;
  }
}

// ---------------- QKV GEMM: xb[b] @ Wq^T + bias -> q,k,v [B,H,NPAD,64] bf16 ------------
__global__ void gemm_qkv_kernel(const __bf16* __restrict__ xb, const __bf16* __restrict__ Wq,
                                const float* __restrict__ bias, __bf16* __restrict__ q,
                                __bf16* __restrict__ k, __bf16* __restrict__ v) {
  __shared__ __align__(16) __bf16 As[2][64 * LDG];
  __shared__ __align__(16) __bf16 Bs[2][64 * LDG];
  int lane = threadIdx.x & 31, wv = threadIdx.x >> 5, tid = threadIdx.x;
  int b = blockIdx.z;
  int mb = blockIdx.x * 64, m0 = mb + wv * 16;
  int n0 = blockIdx.y * 64;
  int hi = lane >> 4, c16 = lane & 15;

  v8f acc[4];
#pragma unroll
  for (int s = 0; s < 4; ++s) {
    float bv = bias[n0 + s * 16 + c16];
#pragma unroll
    for (int r = 0; r < 8; ++r) acc[s][r] = bv;
  }
  gemm_core(xb + (size_t)b * NPAD * CC + (size_t)mb * CC, Wq + (size_t)n0 * CC,
            lane, wv, tid, As, Bs, acc);

#pragma unroll
  for (int s = 0; s < 4; ++s) {
    int j = n0 + s * 16;
    int sel = j >> 10, h = (j >> 6) & 15, d = (j & 63) + c16;
    __bf16* dst = (sel == 0) ? q : (sel == 1) ? k : v;
    __bf16* base = dst + (((size_t)b * HH + h) * NPAD) * HD + d;
#pragma unroll
    for (int r = 0; r < 8; ++r) {
      int row = m0 + r + 8 * hi;
      base[(size_t)row * HD] = (__bf16)acc[s][r];
    }
  }
}

// ---------------- projection GEMM: g[b] @ Wp^T + bias -> out (N,B,C) fp32 --------------
__global__ void gemm_proj_kernel(const __bf16* __restrict__ g, const __bf16* __restrict__ Wp,
                                 const float* __restrict__ bias, float* __restrict__ outb) {
  __shared__ __align__(16) __bf16 As[2][64 * LDG];
  __shared__ __align__(16) __bf16 Bs[2][64 * LDG];
  int lane = threadIdx.x & 31, wv = threadIdx.x >> 5, tid = threadIdx.x;
  int b = blockIdx.z;
  int mb = blockIdx.x * 64, m0 = mb + wv * 16;
  int n0 = blockIdx.y * 64;
  int hi = lane >> 4, c16 = lane & 15;

  v8f acc[4];
#pragma unroll
  for (int s = 0; s < 4; ++s) {
    float bv = bias[n0 + s * 16 + c16];
#pragma unroll
    for (int r = 0; r < 8; ++r) acc[s][r] = bv;
  }
  gemm_core(g + (size_t)b * NPAD * CC + (size_t)mb * CC, Wp + (size_t)n0 * CC,
            lane, wv, tid, As, Bs, acc);

#pragma unroll
  for (int s = 0; s < 4; ++s) {
    int col = n0 + s * 16 + c16;
#pragma unroll
    for (int r = 0; r < 8; ++r) {
      int row = m0 + r + 8 * hi;
      if (row < NN) outb[((size_t)row * BQ + b) * CC + col] = acc[s][r];
    }
  }
}

// ---------------- flash attention: out = softmax(Xq Xk^T * t) @ Y  (fp32 out) ----------
// Double-buffered LDS tiles filled with GLOBAL_LOAD_ASYNC_TO_LDS_B128 (ASYNCcnt).
__global__ void flash_kernel(const __bf16* __restrict__ Xq, const __bf16* __restrict__ Xk,
                             const __bf16* __restrict__ Y, float* __restrict__ out,
                             const float* __restrict__ tbuf, float tconst, int useT,
                             int accum) {
  __shared__ __align__(16) __bf16 Kl[2][32 * LDK];  // K tiles, padded rows
  __shared__ __align__(16) __bf16 Vl[2][32 * LDK];  // V tiles, padded rows
  __shared__ __align__(16) __bf16 Pl[4][16 * 40];   // per-wave P tiles, padded rows
  int lane = threadIdx.x & 31, wv = threadIdx.x >> 5, tid = threadIdx.x;
  int b = blockIdx.z, h = blockIdx.y;
  size_t hoff = ((size_t)b * HH + h) * NPAD * HD;
  const __bf16* Qh = Xq + hoff;
  const __bf16* Kh = Xk + hoff;
  const __bf16* Vh = Y + hoff;
  float t = useT ? tbuf[b] : tconst;
  int m0 = blockIdx.x * 64 + wv * 16;
  int hi = lane >> 4, c16 = lane & 15;

  unsigned kOff[2], vOff[2];
#pragma unroll
  for (int i = 0; i < 2; ++i) {
    kOff[i] = (unsigned)(size_t)(void*)&Kl[i][0];
    vOff[i] = (unsigned)(size_t)(void*)&Vl[i][0];
  }
  // stage tile kt into buffer p: 32 rows x 128B, 8 chunks/row, 2 chunks/thread each tensor
  auto stage = [&](int p, int kt) {
    const char* kg = (const char*)(Kh + (size_t)kt * 32 * HD);
    const char* vg = (const char*)(Vh + (size_t)kt * 32 * HD);
#pragma unroll
    for (int i = 0; i < 2; ++i) {
      int c = tid * 2 + i;  // 0..255
      unsigned row = (unsigned)(c >> 3), seg = (unsigned)(c & 7);
      unsigned loff = row * (LDK * 2) + seg * 16;
      size_t goff = (size_t)row * (HD * 2) + seg * 16;
      async_cp16(kOff[p] + loff, kg + goff);
      async_cp16(vOff[p] + loff, vg + goff);
    }
  };

  v16bf qf0 = ldA(Qh + (size_t)m0 * HD, HD, lane);
  v16bf qf1 = ldA(Qh + (size_t)m0 * HD + 32, HD, lane);

  v8f o[4];
#pragma unroll
  for (int j = 0; j < 4; ++j) o[j] = v8zero();
  float mr[8], lr[8];
#pragma unroll
  for (int r = 0; r < 8; ++r) { mr[r] = -3.0e38f; lr[r] = 0.0f; }

  __bf16* Pw = &Pl[wv][0];
  const int NT = NPAD / 32;

  stage(0, 0);
  wait_async0();
  __syncthreads();

  int p = 0;
  for (int kt = 0; kt < NT; ++kt) {
    if (kt + 1 < NT) stage(1 - p, kt + 1);  // prefetch next tile while computing

    const __bf16* Kp = Kl[p];
    const __bf16* Vp = Vl[p];

    // S tiles: 16 rows x 32 keys (two 16-col subtiles, k-dim 64 in two halves)
    v8f s0 = wmma_bf(qf0, ldB(Kp, LDK, lane), v8zero());
    s0 = wmma_bf(qf1, ldB(Kp + 32, LDK, lane), s0);
    v8f s1 = wmma_bf(qf0, ldB(Kp + 16 * LDK, LDK, lane), v8zero());
    s1 = wmma_bf(qf1, ldB(Kp + 16 * LDK + 32, LDK, lane), s1);

    int key0 = kt * 32 + c16, key1 = key0 + 16;
    bool ok0 = key0 < NN, ok1 = key1 < NN;
#pragma unroll
    for (int r = 0; r < 8; ++r) {
      s0[r] = ok0 ? s0[r] * t : -3.0e38f;
      s1[r] = ok1 ? s1[r] * t : -3.0e38f;
    }
#pragma unroll
    for (int r = 0; r < 8; ++r) {
      float cm = rmax16(fmaxf(s0[r], s1[r]));
      float mn = fmaxf(mr[r], cm);
      float corr = __expf(mr[r] - mn);
      mr[r] = mn;
      float p0 = __expf(s0[r] - mn);
      float p1 = __expf(s1[r] - mn);
      lr[r] = lr[r] * corr + rsum16(p0 + p1);
#pragma unroll
      for (int j = 0; j < 4; ++j) o[j][r] *= corr;
      int prow = r + 8 * hi;
      Pw[prow * 40 + c16] = (__bf16)p0;
      Pw[prow * 40 + 16 + c16] = (__bf16)p1;
    }
    // O += P @ Vtile  (A = P 16x32, B gathered from linear V tile)
    v16bf pa = ldA(Pw, 40, lane);
#pragma unroll
    for (int j = 0; j < 4; ++j)
      o[j] = wmma_bf(pa, ldBVlin(Vp, lane, j), o[j]);

    wait_async0();   // next tile landed
    __syncthreads(); // everyone done reading buffer p before it is refilled
    p ^= 1;
  }

  float* ob = out + hoff;
#pragma unroll
  for (int j = 0; j < 4; ++j) {
    int col = j * 16 + c16;
#pragma unroll
    for (int r = 0; r < 8; ++r) {
      int row = m0 + r + 8 * hi;
      float w = (row < NN) ? o[j][r] / lr[r] : 0.0f;
      size_t idx = (size_t)row * HD + col;
      ob[idx] = accum ? (ob[idx] + w) : w;
    }
  }
}

// ---------------- row-wise L2 normalize [rows][64] -> bf16 ----------------
template <typename T>
__global__ void l2norm_kernel(const T* __restrict__ in, __bf16* __restrict__ outp) {
  size_t row = blockIdx.x;
  int lane = threadIdx.x;
  const T* p = in + row * HD;
  float a = (float)p[lane], c = (float)p[lane + 32];
  float ss = a * a + c * c;
#pragma unroll
  for (int m = 1; m < 32; m <<= 1) ss += __shfl_xor(ss, m, 32);
  float inv = 1.0f / fmaxf(sqrtf(ss), 1e-12f);
  __bf16* q = outp + row * HD;
  q[lane] = (__bf16)(a * inv);
  q[lane + 32] = (__bf16)(c * inv);
}

// ---------------- [B,H,NPAD,64] fp32 -> [B,NPAD,C] bf16, scaled ----------------
__global__ void of2g_kernel(const float* __restrict__ Of, __bf16* __restrict__ g, float scl) {
  const size_t total = (size_t)BQ * NPAD * CC;
  for (size_t i = (size_t)blockIdx.x * blockDim.x + threadIdx.x; i < total;
       i += (size_t)gridDim.x * blockDim.x) {
    int j = (int)(i & (CC - 1));
    size_t bn = i >> 10;
    int n = (int)(bn % NPAD);
    int b = (int)(bn / NPAD);
    int h = j >> 6, d = j & 63;
    float v = (n < NN) ? Of[(((size_t)b * HH + h) * NPAD + n) * HD + d] * scl : 0.0f;
    g[i] = (__bf16)v;
  }
}

// ---------------- host ----------------
extern "C" void kernel_launch(void* const* d_in, const int* in_sizes, int n_in,
                              void* d_out, int out_size, void* d_ws, size_t ws_size,
                              hipStream_t stream) {
  (void)in_sizes; (void)n_in; (void)out_size; (void)ws_size;
  const float* x      = (const float*)d_in[0];
  const float* qkv_w  = (const float*)d_in[1];
  const float* qkv_b  = (const float*)d_in[2];
  const float* proj_w = (const float*)d_in[3];
  const float* proj_b = (const float*)d_in[4];
  float* out_gem = (float*)d_out;
  float* out_ori = out_gem + (size_t)NN * BQ * CC;

  char* w = (char*)d_ws;
  auto alloc = [&](size_t bytes) -> void* {
    void* p = (void*)w;
    w += (bytes + 255) & ~(size_t)255;
    return p;
  };
  const size_t HE = (size_t)BQ * HH * NPAD * HD;  // elems per head-tensor
  __bf16* xb = (__bf16*)alloc((size_t)BQ * NPAD * CC * 2);
  __bf16* Wq = (__bf16*)alloc((size_t)3 * CC * CC * 2);
  __bf16* Wp = (__bf16*)alloc((size_t)CC * CC * 2);
  __bf16* q  = (__bf16*)alloc(HE * 2);
  __bf16* k  = (__bf16*)alloc(HE * 2);
  __bf16* v  = (__bf16*)alloc(HE * 2);
  __bf16* n1 = (__bf16*)alloc(HE * 2);
  __bf16* n2 = (__bf16*)alloc(HE * 2);
  __bf16* n3 = (__bf16*)alloc(HE * 2);
  float*  Zf = (float*)alloc(HE * 4);
  float*  Of = (float*)alloc(HE * 4);
  __bf16* g  = (__bf16*)alloc((size_t)BQ * NPAD * CC * 2);
  float*  rnorm = (float*)alloc((size_t)BQ * NN * 4);
  float*  itemp = (float*)alloc(64);

  const int ROWS = BQ * HH * NPAD;
  dim3 gAttn(NPAD / 64, HH, BQ);

  prep_kernel<<<dim3(NPAD, BQ), 256, 0, stream>>>(x, xb, rnorm);
  fin_kernel<<<BQ, 256, 0, stream>>>(rnorm, itemp);
  convw_kernel<<<2048, 256, 0, stream>>>(qkv_w, proj_w, Wq, Wp);
  gemm_qkv_kernel<<<dim3(NPAD / 64, (3 * CC) / 64, BQ), 128, 0, stream>>>(xb, Wq, qkv_b,
                                                                          q, k, v);
  // ---- ordinary attention branch ----
  flash_kernel<<<gAttn, 128, 0, stream>>>(q, k, v, Of, itemp, SCALE, 0, 0);
  of2g_kernel<<<4096, 256, 0, stream>>>(Of, g, 1.0f);
  gemm_proj_kernel<<<dim3(NPAD / 64, CC / 64, BQ), 128, 0, stream>>>(g, Wp, proj_b, out_ori);

  // ---- self-self branch ----
  l2norm_kernel<__bf16><<<ROWS, 32, 0, stream>>>(v, n1);
  l2norm_kernel<__bf16><<<ROWS, 32, 0, stream>>>(k, n2);
  l2norm_kernel<__bf16><<<ROWS, 32, 0, stream>>>(q, n3);

  flash_kernel<<<gAttn, 128, 0, stream>>>(n1, n1, n1, Zf, itemp, 0.0f, 1, 0);
  l2norm_kernel<float><<<ROWS, 32, 0, stream>>>(Zf, n1);
  flash_kernel<<<gAttn, 128, 0, stream>>>(n2, n2, n2, Zf, itemp, 0.0f, 1, 0);
  l2norm_kernel<float><<<ROWS, 32, 0, stream>>>(Zf, n2);
  flash_kernel<<<gAttn, 128, 0, stream>>>(n3, n3, n3, Zf, itemp, 0.0f, 1, 0);
  l2norm_kernel<float><<<ROWS, 32, 0, stream>>>(Zf, n3);

  flash_kernel<<<gAttn, 128, 0, stream>>>(n1, n1, v, Of, itemp, 0.0f, 1, 0);
  flash_kernel<<<gAttn, 128, 0, stream>>>(n2, n2, v, Of, itemp, 0.0f, 1, 1);
  flash_kernel<<<gAttn, 128, 0, stream>>>(n3, n3, v, Of, itemp, 0.0f, 1, 1);

  of2g_kernel<<<4096, 256, 0, stream>>>(Of, g, 1.0f / 3.0f);
  gemm_proj_kernel<<<dim3(NPAD / 64, CC / 64, BQ), 128, 0, stream>>>(g, Wp, proj_b, out_gem);
}
